// FCOS_72825465471345
// MI455X (gfx1250) — matrix operand order repaired and checked by reading, not compiled, not run on hardware
//
#include <hip/hip_runtime.h>
#include <hip/hip_bf16.h>
#include <stdint.h>

typedef __attribute__((ext_vector_type(16))) _Float16 v16h;
typedef __attribute__((ext_vector_type(8)))  _Float16 v8h;
typedef __attribute__((ext_vector_type(8)))  float    v8f;

#define KSLAB 256
#define LDSW  (KSLAB + 8)     // half-element LDS row stride (16B aligned, bank skew)
#define LBUF  (64 * LDSW)     // halves per slab buffer

// Set to 0 to fall back to plain global->VGPR->LDS staging.
#define CONV_ASYNC_LDS 1

// ---------------------------------------------------------------------------
// Implicit-GEMM conv, NHWC fp16, f32 accumulate via v_wmma_f32_16x16x32_f16.
// Block = 256 threads = 8 waves. Block tile: M=256 (32 rows/wave), N=64.
// Per K-chunk (32) per wave: 2 A-frag global loads (4x b128), 4 B-frags from
// LDS (8x ds_read_b128), 8 WMMAs. Weights stream through a double-buffered
// pair of 32KB LDS slabs via global_load_async_to_lds_b128: slab s+1 is
// prefetched while slab s is consumed; ASYNCcnt loads retire in order, so
// "wait ASYNCcnt <= (loads issued for s+1)" guarantees slab s has landed.
// M = B*H*W must be a multiple of 256 (true for 2048 / 8192 / 32768).
// ---------------------------------------------------------------------------
template<bool RELU, bool F16OUT>
__global__ __launch_bounds__(256, 2)
void conv_wmma(const _Float16* __restrict__ X,
               const _Float16* __restrict__ Wp,
               const float*    __restrict__ Bp,
               void*           __restrict__ Yv,
               int H, int Wd, int Cin, int KS, int Cout)
{
    __shared__ __align__(16) _Float16 lb[2 * LBUF];

    const int Ktot = KS * KS * Cin;
    const int HW   = H * Wd;
    const int tid  = threadIdx.x;
    const int wave = tid >> 5;
    const int lane = tid & 31;
    const int l16  = lane & 15;
    const int lhi  = lane >> 4;          // 0 or 1

    const int m0 = blockIdx.x * 256 + wave * 32;   // wave's 32 output rows
    const int n0 = blockIdx.y * 64;                // block's 64 output cols

    // Two per-lane row descriptors (rows m0+l16 and m0+16+l16).
    int py[2], px[2];
    long rowbase[2];
    #pragma unroll
    for (int t = 0; t < 2; ++t) {
        const int m   = m0 + t * 16 + l16;
        const int bi  = m / HW;
        const int rem = m - bi * HW;
        py[t] = rem / Wd;
        px[t] = rem - py[t] * Wd;
        rowbase[t] = ((long)(bi * H + py[t]) * Wd + px[t]) * Cin;  // center tap
    }

    // Issue one weight slab (64 rows x slab halves) into LDS buffer `buf`.
    // Returns the number of async loads issued per thread (uniform).
    auto issue_slab = [&](int kb, int buf) -> int {
        const int slab = (Ktot - kb < KSLAB) ? (Ktot - kb) : KSLAB;  // mult of 32
        const int rowc = slab >> 3;                                  // 16B chunks/row
        for (int q = tid; q < 64 * rowc; q += 256) {                 // uniform trips
            const int n  = q / rowc;
            const int k8 = (q - n * rowc) << 3;
            const _Float16* gp = &Wp[(size_t)(n0 + n) * Ktot + kb + k8];
#if CONV_ASYNC_LDS
            const unsigned lds = (unsigned)(uintptr_t)&lb[buf * LBUF + n * LDSW + k8];
            asm volatile("global_load_async_to_lds_b128 %0, %1, off"
                         :: "v"(lds), "v"((unsigned long long)(uintptr_t)gp)
                         : "memory");
#else
            *(v8h*)&lb[buf * LBUF + n * LDSW + k8] = *(const v8h*)gp;
#endif
        }
        return rowc >> 2;   // (64*rowc)/256
    };

    const v16h zero16 = {};
    v8f accA[4] = {};
    v8f accB[4] = {};

    // Loop-carried tap state: K index = (ky*KS+kx)*Cin + cb, all multiples of 32.
    int cb = 0, ky = 0, kx = 0;
    const int pad   = (KS == 3) ? 1 : 0;
    const int nslab = (Ktot + KSLAB - 1) / KSLAB;

    issue_slab(0, 0);

    for (int s = 0; s < nslab; ++s) {
        __syncthreads();   // all waves done reading buf[(s+1)&1] (slab s-1 compute)

        int nxt = 0;
        if (s + 1 < nslab) nxt = issue_slab((s + 1) * KSLAB, (s + 1) & 1);

#if CONV_ASYNC_LDS
        // Slab-s loads were issued earlier; async loads retire in order, so
        // draining down to the just-issued count means slab s has landed.
        if (nxt >= 8)      asm volatile("s_wait_asynccnt 0x8" ::: "memory");
        else if (nxt == 5) asm volatile("s_wait_asynccnt 0x5" ::: "memory");
        else if (nxt == 2) asm volatile("s_wait_asynccnt 0x2" ::: "memory");
        else               asm volatile("s_wait_asynccnt 0x0" ::: "memory");
#endif
        __syncthreads();   // slab s visible to every wave

        const _Float16* lbp = lb + (s & 1) * LBUF;
        const int kbase = s * KSLAB;
        const int slab  = (Ktot - kbase < KSLAB) ? (Ktot - kbase) : KSLAB;

        for (int kc = 0; kc < slab; kc += 32) {
            const int  dy   = ky - pad;
            const int  dx   = kx - pad;
            const long toff = ((long)dy * Wd + dx) * Cin + cb + lhi * 8;

            union { v16h v; v8h h[2]; } a0, a1;
            {
                const int sy = py[0] + dy, sx = px[0] + dx;
                if ((unsigned)sy < (unsigned)H && (unsigned)sx < (unsigned)Wd) {
                    const _Float16* sp = X + rowbase[0] + toff;
                    a0.h[0] = *(const v8h*)(sp);
                    a0.h[1] = *(const v8h*)(sp + 16);
                } else a0.v = zero16;
            }
            {
                const int sy = py[1] + dy, sx = px[1] + dx;
                if ((unsigned)sy < (unsigned)H && (unsigned)sx < (unsigned)Wd) {
                    const _Float16* sp = X + rowbase[1] + toff;
                    a1.h[0] = *(const v8h*)(sp);
                    a1.h[1] = *(const v8h*)(sp + 16);
                } else a1.v = zero16;
            }

            // 4 B fragments, each feeding two WMMAs (A reuse across M).
            #pragma unroll
            for (int j = 0; j < 4; ++j) {
                union { v16h v; v8h h[2]; } b;
                const _Float16* lr = &lbp[(j * 16 + l16) * LDSW + kc + lhi * 8];
                b.h[0] = *(const v8h*)(lr);
                b.h[1] = *(const v8h*)(lr + 16);
                accA[j] = __builtin_amdgcn_wmma_f32_16x16x32_f16(
                    false, a0.v, false, b.v, (short)0, accA[j], false, false);
                accB[j] = __builtin_amdgcn_wmma_f32_16x16x32_f16(
                    false, a1.v, false, b.v, (short)0, accB[j], false, false);
            }

            // advance tap state (no divisions)
            cb += 32;
            if (cb == Cin) { cb = 0; ++kx; if (kx == KS) { kx = 0; ++ky; } }
        }
    }

    // ---- epilogue: lane owns column n; rows m0 + t*16 + r + 8*lhi ----
    #pragma unroll
    for (int j = 0; j < 4; ++j) {
        const int   n  = n0 + j * 16 + l16;
        const float bv = Bp[n];                      // bias padded to 256 -> safe
        if (n < Cout) {
            #pragma unroll
            for (int t = 0; t < 2; ++t) {
                const v8f acc = t ? accB[j] : accA[j];
                #pragma unroll
                for (int r = 0; r < 8; ++r) {
                    float v = acc[r] + bv;
                    if (RELU) v = v > 0.f ? v : 0.f;
                    const size_t idx =
                        (size_t)(m0 + t * 16 + r + 8 * lhi) * Cout + n;
                    if (F16OUT) ((_Float16*)Yv)[idx] = (_Float16)v;
                    else        ((float*)Yv)[idx]    = v;
                }
            }
        }
    }
}

// --------------------------- utility kernels -------------------------------

__global__ void k_to_nhwc(const float* __restrict__ src, _Float16* __restrict__ dst,
                          int C, int H, int W, int Cp, int total)
{
    int idx = blockIdx.x * 256 + threadIdx.x;
    if (idx >= total) return;
    int c = idx % Cp; int s = idx / Cp;
    int x = s % W;  s /= W;
    int y = s % H;  int b = s / H;
    float v = (c < C) ? src[(((size_t)b * C + c) * H + y) * W + x] : 0.f;
    dst[idx] = (_Float16)v;
}

__global__ void k_pack_w(const float* __restrict__ src, _Float16* __restrict__ dst,
                         int Cout, int Cin, int KS, int Cinp, int total)
{
    int idx = blockIdx.x * 256 + threadIdx.x;
    if (idx >= total) return;
    const int kperN = KS * KS * Cinp;
    int n = idx / kperN; int r = idx - n * kperN;
    int plane = r / Cinp; int ci = r - plane * Cinp;
    int ky = plane / KS;  int kx = plane - ky * KS;
    float v = 0.f;
    if (n < Cout && ci < Cin)
        v = src[(((size_t)n * Cin + ci) * KS + ky) * KS + kx];
    dst[idx] = (_Float16)v;
}

__global__ void k_pack_b(const float* __restrict__ src, float* __restrict__ dst, int Cout)
{
    int idx = threadIdx.x;            // one 256-thread block
    dst[idx] = (idx < Cout) ? src[idx] : 0.f;
}

__global__ void k_ups_add(const _Float16* __restrict__ lat, const _Float16* __restrict__ up,
                          _Float16* __restrict__ out, int H, int W, int C, int total)
{
    int idx = blockIdx.x * 256 + threadIdx.x;
    if (idx >= total) return;
    int c = idx % C; int s = idx / C;
    int x = s % W;  s /= W;
    int y = s % H;  int b = s / H;
    _Float16 u = up[(((size_t)b * (H / 2) + (y >> 1)) * (W / 2) + (x >> 1)) * C + c];
    out[idx] = (_Float16)(lat[idx] + u);
}

// ------------------------------ host side ----------------------------------

static inline void run_conv(hipStream_t st, const _Float16* X, const _Float16* Wp,
                            const float* Bp, void* Y, int M, int H, int W,
                            int Cin, int KS, int Cout, bool relu, bool f16out)
{
    dim3 g(M / 256, (Cout + 63) / 64), b(256);
    if (relu)        conv_wmma<true , true ><<<g, b, 0, st>>>(X, Wp, Bp, Y, H, W, Cin, KS, Cout);
    else if (f16out) conv_wmma<false, true ><<<g, b, 0, st>>>(X, Wp, Bp, Y, H, W, Cin, KS, Cout);
    else             conv_wmma<false, false><<<g, b, 0, st>>>(X, Wp, Bp, Y, H, W, Cin, KS, Cout);
}

extern "C" void kernel_launch(void* const* d_in, const int* in_sizes, int n_in,
                              void* d_out, int out_size, void* d_ws, size_t ws_size,
                              hipStream_t stream)
{
    (void)n_in; (void)out_size; (void)ws_size;

    // ---- parameter index maps (insertion-order vs pytree-sorted flattening) ----
    int iLW[3], iLB[3], iOW[3], iOB[3], iSCW[4], iSCB[4], iSBW[4], iSBB[4];
    int iPCW, iPCB, iPBW, iPBB, iPTW, iPTB;
    const bool sorted = (in_sizes[3] == 256);   // lat_c3_b first => sorted-key order
    if (!sorted) {
        for (int i = 0; i < 3; ++i) { iLW[i] = 3 + 2 * i; iLB[i] = 4 + 2 * i; }
        for (int i = 0; i < 3; ++i) { iOW[i] = 9 + 2 * i; iOB[i] = 10 + 2 * i; }
        for (int i = 0; i < 4; ++i) { iSCW[i] = 15 + 2 * i; iSCB[i] = 16 + 2 * i; }
        for (int i = 0; i < 4; ++i) { iSBW[i] = 23 + 2 * i; iSBB[i] = 24 + 2 * i; }
        iPCW = 31; iPCB = 32; iPBW = 33; iPBB = 34; iPTW = 35; iPTB = 36;
    } else {
        for (int i = 0; i < 3; ++i) { iLB[i] = 3 + 2 * i; iLW[i] = 4 + 2 * i; }
        for (int i = 0; i < 3; ++i) { iOB[i] = 9 + 2 * i; iOW[i] = 10 + 2 * i; }
        iPBB = 15; iPBW = 16; iPCB = 17; iPCW = 18; iPTB = 19; iPTW = 20;
        for (int i = 0; i < 4; ++i) { iSBB[i] = 21 + 2 * i; iSBW[i] = 22 + 2 * i; }
        for (int i = 0; i < 4; ++i) { iSCB[i] = 29 + 2 * i; iSCW[i] = 30 + 2 * i; }
    }

    // ---- workspace bump allocator ----
    uintptr_t cur = (uintptr_t)d_ws;
    auto alloc = [&](size_t bytes) -> void* {
        cur = (cur + 255) & ~(uintptr_t)255;
        void* p = (void*)cur; cur += bytes; return p;
    };
    auto ah = [&](size_t n) { return (_Float16*)alloc(n * sizeof(_Float16)); };
    auto af = [&](size_t n) { return (float*)alloc(n * sizeof(float)); };

    // activations (NHWC fp16)
    _Float16* x3   = ah((size_t)8 * 64 * 64 * 64);
    _Float16* x4   = ah((size_t)8 * 32 * 32 * 160);
    _Float16* x5   = ah((size_t)8 * 16 * 16 * 416);   // 400 -> padded 416
    _Float16* lat5 = ah((size_t)2048 * 256);
    _Float16* p5b  = ah((size_t)2048 * 256);
    _Float16* lat4 = ah((size_t)8192 * 256);
    _Float16* t4   = ah((size_t)8192 * 256);
    _Float16* p4b  = ah((size_t)8192 * 256);
    _Float16* lat3 = ah((size_t)32768 * 256);
    _Float16* t3   = ah((size_t)32768 * 256);
    _Float16* p3b  = ah((size_t)32768 * 256);
    _Float16* sA   = ah((size_t)32768 * 256);
    _Float16* sB   = ah((size_t)32768 * 256);

    // packed weights (fp16) + padded biases (fp32, 256 each)
    _Float16* wlat[3] = { ah(256 * 64), ah(256 * 160), ah(256 * 416) };
    _Float16* wout[3] = { ah((size_t)256 * 2304), ah((size_t)256 * 2304), ah((size_t)256 * 2304) };
    _Float16 *wsc[4], *wsb[4];
    for (int i = 0; i < 4; ++i) wsc[i] = ah((size_t)256 * 2304);
    for (int i = 0; i < 4; ++i) wsb[i] = ah((size_t)256 * 2304);
    _Float16* wpc = ah((size_t)128 * 2304);
    _Float16* wpb = ah((size_t)64 * 2304);
    _Float16* wpt = ah((size_t)64 * 2304);
    float *blat[3], *bout[3], *bsc[4], *bsb[4];
    for (int i = 0; i < 3; ++i) blat[i] = af(256);
    for (int i = 0; i < 3; ++i) bout[i] = af(256);
    for (int i = 0; i < 4; ++i) bsc[i]  = af(256);
    for (int i = 0; i < 4; ++i) bsb[i]  = af(256);
    float* bpc = af(256); float* bpb = af(256); float* bpt = af(256);

    auto packw = [&](int src, _Float16* dst, int Cout, int Cin, int KS, int Cinp, int Np) {
        int total = Np * KS * KS * Cinp;
        k_pack_w<<<(total + 255) / 256, 256, 0, stream>>>(
            (const float*)d_in[src], dst, Cout, Cin, KS, Cinp, total);
    };
    auto packb = [&](int src, float* dst, int Cout) {
        k_pack_b<<<1, 256, 0, stream>>>((const float*)d_in[src], dst, Cout);
    };

    // ---- input conversion NCHW f32 -> NHWC f16 ----
    {
        int t3n = 8 * 64 * 64 * 64;
        k_to_nhwc<<<(t3n + 255) / 256, 256, 0, stream>>>((const float*)d_in[0], x3, 64, 64, 64, 64, t3n);
        int t4n = 8 * 32 * 32 * 160;
        k_to_nhwc<<<(t4n + 255) / 256, 256, 0, stream>>>((const float*)d_in[1], x4, 160, 32, 32, 160, t4n);
        int t5n = 8 * 16 * 16 * 416;
        k_to_nhwc<<<(t5n + 255) / 256, 256, 0, stream>>>((const float*)d_in[2], x5, 400, 16, 16, 416, t5n);
    }

    // ---- weight / bias packing ----
    packw(iLW[0], wlat[0], 256, 64, 1, 64, 256);   packb(iLB[0], blat[0], 256);
    packw(iLW[1], wlat[1], 256, 160, 1, 160, 256); packb(iLB[1], blat[1], 256);
    packw(iLW[2], wlat[2], 256, 400, 1, 416, 256); packb(iLB[2], blat[2], 256);
    for (int i = 0; i < 3; ++i) { packw(iOW[i], wout[i], 256, 256, 3, 256, 256); packb(iOB[i], bout[i], 256); }
    for (int i = 0; i < 4; ++i) { packw(iSCW[i], wsc[i], 256, 256, 3, 256, 256); packb(iSCB[i], bsc[i], 256); }
    for (int i = 0; i < 4; ++i) { packw(iSBW[i], wsb[i], 256, 256, 3, 256, 256); packb(iSBB[i], bsb[i], 256); }
    packw(iPCW, wpc, 80, 256, 3, 256, 128); packb(iPCB, bpc, 80);
    packw(iPBW, wpb, 4, 256, 3, 256, 64);   packb(iPBB, bpb, 4);
    packw(iPTW, wpt, 1, 256, 3, 256, 64);   packb(iPTB, bpt, 1);

    // ---- FPN ----
    run_conv(stream, x5,   wlat[2], blat[2], lat5, 2048, 16, 16, 416, 1, 256, false, true);
    run_conv(stream, lat5, wout[2], bout[2], p5b,  2048, 16, 16, 256, 3, 256, false, true);
    run_conv(stream, x4,   wlat[1], blat[1], lat4, 8192, 32, 32, 160, 1, 256, false, true);
    { int t = 8192 * 256; k_ups_add<<<(t + 255) / 256, 256, 0, stream>>>(lat4, p5b, t4, 32, 32, 256, t); }
    run_conv(stream, t4,   wout[1], bout[1], p4b,  8192, 32, 32, 256, 3, 256, false, true);
    run_conv(stream, x3,   wlat[0], blat[0], lat3, 32768, 64, 64, 64, 1, 256, false, true);
    { int t = 32768 * 256; k_ups_add<<<(t + 255) / 256, 256, 0, stream>>>(lat3, p4b, t3, 64, 64, 256, t); }
    run_conv(stream, t3,   wout[0], bout[0], p3b,  32768, 64, 64, 256, 3, 256, false, true);

    // ---- output slice offsets (fp32, (B, H*W, C), return order) ----
    float* out = (float*)d_out;
    size_t offc[3], offb[3], offt[3];
    offc[0] = 0;
    offc[1] = offc[0] + (size_t)8 * 4096 * 80;
    offc[2] = offc[1] + (size_t)8 * 1024 * 80;
    offb[0] = offc[2] + (size_t)8 * 256 * 80;
    offb[1] = offb[0] + (size_t)8 * 4096 * 4;
    offb[2] = offb[1] + (size_t)8 * 1024 * 4;
    offt[0] = offb[2] + (size_t)8 * 256 * 4;
    offt[1] = offt[0] + (size_t)8 * 4096;
    offt[2] = offt[1] + (size_t)8 * 1024;

    // ---- FCOS head: stems + prediction convs per level ----
    const _Float16* feats[3] = { p3b, p4b, p5b };
    const int Hs[3] = { 64, 32, 16 };
    const int Ms[3] = { 32768, 8192, 2048 };

    for (int L = 0; L < 3; ++L) {
        const int H = Hs[L], M = Ms[L];
        // cls branch
        run_conv(stream, feats[L], wsc[0], bsc[0], sA, M, H, H, 256, 3, 256, true, true);
        run_conv(stream, sA,       wsc[1], bsc[1], sB, M, H, H, 256, 3, 256, true, true);
        run_conv(stream, sB,       wsc[2], bsc[2], sA, M, H, H, 256, 3, 256, true, true);
        run_conv(stream, sA,       wsc[3], bsc[3], sB, M, H, H, 256, 3, 256, true, true);
        run_conv(stream, sB, wpc, bpc, out + offc[L], M, H, H, 256, 3, 80, false, false);
        // box branch
        run_conv(stream, feats[L], wsb[0], bsb[0], sA, M, H, H, 256, 3, 256, true, true);
        run_conv(stream, sA,       wsb[1], bsb[1], sB, M, H, H, 256, 3, 256, true, true);
        run_conv(stream, sB,       wsb[2], bsb[2], sA, M, H, H, 256, 3, 256, true, true);
        run_conv(stream, sA,       wsb[3], bsb[3], sB, M, H, H, 256, 3, 256, true, true);
        run_conv(stream, sB, wpb, bpb, out + offb[L], M, H, H, 256, 3, 4, false, false);
        run_conv(stream, sB, wpt, bpt, out + offt[L], M, H, H, 256, 3, 1, false, false);
    }
}
